// MiST_45251775430849
// MI455X (gfx1250) — matrix operand order repaired and checked by reading, not compiled, not run on hardware
//
#include <hip/hip_runtime.h>

#define NHID   256
#define GATES  1024
#define TSTEPS 42
#define BTOT   2464
#define BTILES 154    // BTOT / 16
#define NTG    64     // GATES / 16 (n-tiles)
#define KTILES 8      // NHID / 32 (k-tiles of 32 for f16 wmma)

typedef __attribute__((ext_vector_type(16))) _Float16 v16h_t;
typedef __attribute__((ext_vector_type(8)))  _Float16 v8h_t;
typedef __attribute__((ext_vector_type(8)))  float    v8f_t;

union Frag32 { v16h_t v; v8h_t h[2]; };

__device__ __forceinline__ float sigmf(float x) { return 1.0f / (1.0f + __expf(-x)); }

// Plain cached 32-byte fragment load: two global_load_b128, clause-able,
// tracked by LOADcnt so the scheduler can overlap them with WMMA.
__device__ __forceinline__ v16h_t load_frag(const _Float16* p) {
    Frag32 f;
    f.h[0] = *((const v8h_t*)p);
    f.h[1] = *((const v8h_t*)p + 1);
    return f.v;
}

// ---------------------------------------------------------------------------
// Pack Whh (1024x256 f32, gates[b,g]=sum_k h[b,k]*Whh[g,k]) into f16 WMMA
// B-fragments: B tile is 32(K) x 16(N). Per lane: lane<16 -> col=lane,
// K = kt*32 + e (e<8), kt*32 + 16 + (e-8) (e>=8); lane>=16 -> col=lane-16,
// K = kt*32 + 8 + e (e<8), kt*32 + 24 + (e-8) (e>=8).
// Storage: [ntG][kt][lane][16] contiguous halves.
// ---------------------------------------------------------------------------
__global__ void k_pack_whh(const float* __restrict__ Whh, _Float16* __restrict__ dst) {
    int idx = blockIdx.x * 256 + threadIdx.x;          // 64*8*32 = 16384 items
    if (idx >= NTG * KTILES * 32) return;
    int ntG  = idx / (KTILES * 32);
    int rem  = idx % (KTILES * 32);
    int kt   = rem / 32;
    int lane = rem % 32;
    int n    = ntG * 16 + (lane & 15);
    int kbase = kt * 32 + ((lane < 16) ? 0 : 8);
    _Float16* p = dst + ((size_t)(ntG * KTILES + kt) * 32 + lane) * 16;
#pragma unroll
    for (int e = 0; e < 16; ++e)
        p[e] = (_Float16)Whh[(size_t)n * NHID + (kbase + ((e < 8) ? e : e + 8))];
}

// Wih (1024x20) -> one padded K=32 B-fragment per n-tile: [ntG][lane][16]
__global__ void k_pack_wih(const float* __restrict__ Wih, _Float16* __restrict__ dst) {
    int idx = blockIdx.x * 256 + threadIdx.x;          // 64*32 = 2048 items
    if (idx >= NTG * 32) return;
    int ntG  = idx / 32;
    int lane = idx % 32;
    int n    = ntG * 16 + (lane & 15);
    int kbase = (lane < 16) ? 0 : 8;
    _Float16* p = dst + ((size_t)ntG * 32 + lane) * 16;
#pragma unroll
    for (int e = 0; e < 16; ++e) {
        int k = kbase + ((e < 8) ? e : e + 8);
        p[e] = (k < 20) ? (_Float16)Wih[(size_t)n * 20 + k] : (_Float16)0.0f;
    }
}

// x slice -> A-fragments (16x32, K padded from 20). ISA 16-bit A 16x32 layout:
// lane<16: m=lane, halves e<8 -> K=e, e>=8 -> K=16+(e-8)
// lane>=16: m=lane-16, halves e<8 -> K=8+e, e>=8 -> K=24+(e-8)
// Storage: [btile][t][lane][16]
__global__ void k_pack_x(const float* __restrict__ x, _Float16* __restrict__ dst) {
    int idx = blockIdx.x * 256 + threadIdx.x;          // 154*42*32 = 206976 items
    if (idx >= BTILES * TSTEPS * 32) return;
    int btile = idx / (TSTEPS * 32);
    int rem   = idx % (TSTEPS * 32);
    int t     = rem / 32;
    int lane  = rem % 32;
    int m     = lane & 15;
    size_t b  = (size_t)btile * 16 + m;
    const float* xrow = x + (b * TSTEPS + t) * 120 + 100;
    _Float16* p = dst + ((size_t)(btile * TSTEPS + t) * 32 + lane) * 16;
#pragma unroll
    for (int e = 0; e < 16; ++e) {
        int k;
        if (lane < 16) k = (e < 8) ? e       : 16 + (e - 8);
        else           k = (e < 8) ? 8 + e   : 24 + (e - 8);
        p[e] = (k < 20) ? (_Float16)xrow[k] : (_Float16)0.0f;
    }
}

// bias = bih + bhh ; wvh = wvW[0,0:256] ; uvc = wr[8].wvW[256:288] + wc[1].wvW[288:320] + wvb
__global__ void k_misc(const float* __restrict__ bih, const float* __restrict__ bhh,
                       const float* __restrict__ wvW, const float* __restrict__ wr,
                       const float* __restrict__ wc,  const float* __restrict__ wvb,
                       float* __restrict__ bias, float* __restrict__ wvh,
                       float* __restrict__ uvc) {
    int gid = blockIdx.x * 256 + threadIdx.x;
    if (gid < GATES)              bias[gid]        = bih[gid] + bhh[gid];
    else if (gid < GATES + NHID)  wvh[gid - GATES] = wvW[gid - GATES];
    if (gid == 0) {
        float s = wvb[0];
        for (int i = 0; i < 32; ++i) s += wr[8 * 32 + i] * wvW[256 + i];
        for (int i = 0; i < 32; ++i) s += wc[1 * 32 + i] * wvW[288 + i];
        uvc[0] = s;
    }
}

// ---------------------------------------------------------------------------
// Fused LSTM + attention partials. One workgroup = 16 batch rows, 8 waves.
// Wave w owns hidden cols [w*32, w*32+32): for each gate gi and half hh, the
// global n-tile is gi*16 + w*2 + hh. 72 v_wmma_f32_16x16x32_f16 per wave/step.
// Whh fragments are streamed from L2 every step (512KB, resident in 192MB L2).
// An empty inline-asm tie on *integer offsets* makes the load addresses
// loop-variant (so LICM cannot hoist 512 VGPRs of fragments out of the t-loop)
// while preserving pointer provenance, keeping the loads as cached
// global_load_b128 (LOADcnt only) rather than flat_load (LOADcnt+DScnt).
// ---------------------------------------------------------------------------
__global__ __launch_bounds__(256) void k_lstm_attn(
    const _Float16* __restrict__ whhP, const _Float16* __restrict__ wihP,
    const _Float16* __restrict__ xP,   const float* __restrict__ bias,
    const float* __restrict__ wvh,     const float* __restrict__ uvc,
    float* __restrict__ esum,          float* __restrict__ hsum)
{
    __shared__ _Float16 hT[16][NHID];   // current hidden state, f16, 8KB
    __shared__ float    eBuf[16];

    const int tid   = threadIdx.x;
    const int wave  = tid >> 5;
    const int lane  = tid & 31;
    const int btile = blockIdx.x;
    const int m     = lane & 15;
    const int koff  = (lane < 16) ? 0 : 8;

    for (int i = tid; i < 16 * NHID; i += 256) ((_Float16*)hT)[i] = (_Float16)0.0f;

    float c[16];                        // cell state: 2 halves x 8 rows per lane
#pragma unroll
    for (int i = 0; i < 16; ++i) c[i] = 0.0f;

    float biasv[8];                     // per n-tile bias for this lane's column
#pragma unroll
    for (int gi = 0; gi < 4; ++gi)
#pragma unroll
        for (int hh = 0; hh < 2; ++hh)
            biasv[gi * 2 + hh] = bias[gi * NHID + wave * 32 + hh * 16 + m];

    __syncthreads();

    size_t wOff = 0, wiOff = 0;   // opaque per-iteration offsets (stay 0)

    for (int t = 0; t < TSTEPS; ++t) {
        // Redefine the offsets each iteration (emits no instructions): blocks
        // LICM of the loop-invariant weight fragment loads without breaking
        // global address-space inference.
        asm volatile("" : "+s"(wOff), "+s"(wiOff));
        const _Float16* wp  = whhP + wOff;
        const _Float16* wip = wihP + wiOff;

        // x-projection A fragment (prepacked, contiguous 32B per lane)
        v16h_t xf = load_frag(xP + ((size_t)(btile * TSTEPS + t) * 32 + lane) * 16);

        // A fragments of h_{t-1} from LDS (two b128 chunks per k-tile)
        v16h_t aF[KTILES];
#pragma unroll
        for (int kt = 0; kt < KTILES; ++kt) {
            Frag32 a;
            a.h[0] = *(const v8h_t*)&hT[m][kt * 32 + koff];
            a.h[1] = *(const v8h_t*)&hT[m][kt * 32 + 16 + koff];
            aF[kt] = a.v;
        }

        v8f_t acc[8];
#pragma unroll
        for (int q = 0; q < 8; ++q) {
            int gi = q >> 1, hh = q & 1;
            int ntG = gi * 16 + wave * 2 + hh;
            v8f_t a0;
#pragma unroll
            for (int e = 0; e < 8; ++e) a0[e] = biasv[q];
            // input projection (K padded to 32)
            v16h_t wf = load_frag(wip + ((size_t)ntG * 32 + lane) * 16);
            a0 = __builtin_amdgcn_wmma_f32_16x16x32_f16(false, xf, false, wf,
                                                        (short)0, a0, false, false);
            // recurrent GEMM: 8 k-tiles of Whh, streamed (L2-resident)
            const _Float16* bp = wp + (size_t)ntG * KTILES * 32 * 16 + (size_t)lane * 16;
#pragma unroll
            for (int kt = 0; kt < KTILES; ++kt) {
                v16h_t bF = load_frag(bp + (size_t)kt * 32 * 16);
                a0 = __builtin_amdgcn_wmma_f32_16x16x32_f16(false, aF[kt], false, bF,
                                                            (short)0, a0, false, false);
            }
            acc[q] = a0;
        }
        __syncthreads();   // all waves done reading h_{t-1}

        // elementwise gate math + write h_t
#pragma unroll
        for (int hh = 0; hh < 2; ++hh) {
            int col = wave * 32 + hh * 16 + m;
#pragma unroll
            for (int r = 0; r < 8; ++r) {
                float iv = acc[0 + hh][r];
                float fv = acc[2 + hh][r];
                float gv = acc[4 + hh][r];
                float ov = acc[6 + hh][r];
                float cc = c[hh * 8 + r];
                cc = sigmf(fv) * cc + sigmf(iv) * tanhf(gv);
                c[hh * 8 + r] = cc;
                float hv = sigmf(ov) * tanhf(cc);
                int row = r + ((lane < 16) ? 0 : 8);
                hT[row][col] = (_Float16)hv;
            }
        }
        __syncthreads();   // h_t visible

        // attention partials: e_m = exp(tanh(h_m . wvh + uvc))
        if (tid < 16) {
            float d = uvc[0];
            for (int k = 0; k < NHID; ++k) d += (float)hT[tid][k] * wvh[k];
            eBuf[tid] = __expf(tanhf(d));
        }
        __syncthreads();

        {
            float s = 0.0f;
#pragma unroll
            for (int mm = 0; mm < 16; ++mm) s += eBuf[mm] * (float)hT[mm][tid];
            hsum[((size_t)btile * TSTEPS + t) * NHID + tid] = s;
            if (tid == 0) {
                float es = 0.0f;
#pragma unroll
                for (int mm = 0; mm < 16; ++mm) es += eBuf[mm];
                esum[(size_t)btile * TSTEPS + t] = es;
            }
        }
        // next iteration's LDS writes are fenced by the post-wmma barrier
    }
}

// ---------------------------------------------------------------------------
// Head: deterministic reduction over batch tiles, softmax-pool, ReLU MLP.
// One block per timestep.
// ---------------------------------------------------------------------------
__global__ __launch_bounds__(256) void k_head(
    const float* __restrict__ esum, const float* __restrict__ hsum,
    const float* __restrict__ W1,   const float* __restrict__ b1,
    const float* __restrict__ W2,   const float* __restrict__ b2,
    float* __restrict__ out)
{
    __shared__ float pooled[NHID];
    __shared__ float z[32];
    __shared__ float denom;
    int t = blockIdx.x, tid = threadIdx.x;

    if (tid == 0) {
        float d = 0.0f;
        for (int bt = 0; bt < BTILES; ++bt) d += esum[(size_t)bt * TSTEPS + t];
        denom = d;
    }
    __syncthreads();
    {
        float s = 0.0f;
        for (int bt = 0; bt < BTILES; ++bt)
            s += hsum[((size_t)bt * TSTEPS + t) * NHID + tid];
        pooled[tid] = s / denom;
    }
    __syncthreads();
    if (tid < 32) {
        float a = b1[tid];
        for (int h = 0; h < NHID; ++h) a += pooled[h] * W1[(size_t)tid * NHID + h];
        z[tid] = fmaxf(a, 0.0f);
    }
    __syncthreads();
    if (tid == 0) {
        float o = b2[0];
        for (int j = 0; j < 32; ++j) o += z[j] * W2[j];
        out[t] = o;
    }
}

// ---------------------------------------------------------------------------
extern "C" void kernel_launch(void* const* d_in, const int* in_sizes, int n_in,
                              void* d_out, int out_size, void* d_ws, size_t ws_size,
                              hipStream_t stream) {
    const float* x   = (const float*)d_in[0];
    const float* wr  = (const float*)d_in[1];
    const float* wc  = (const float*)d_in[2];
    const float* Wih = (const float*)d_in[3];
    const float* Whh = (const float*)d_in[4];
    const float* bih = (const float*)d_in[5];
    const float* bhh = (const float*)d_in[6];
    const float* wvW = (const float*)d_in[7];
    const float* wvb = (const float*)d_in[8];
    const float* W1  = (const float*)d_in[9];
    const float* b1  = (const float*)d_in[10];
    const float* W2  = (const float*)d_in[11];
    const float* b2  = (const float*)d_in[12];
    float* out = (float*)d_out;

    char* ws = (char*)d_ws;
    constexpr size_t SZ_WHH  = (size_t)NTG * KTILES * 32 * 16 * 2;          // 524288
    constexpr size_t SZ_WIH  = (size_t)NTG * 32 * 16 * 2;                   // 65536
    constexpr size_t SZ_XP   = (size_t)BTILES * TSTEPS * 32 * 16 * 2;       // 6623232
    constexpr size_t SZ_BIAS = GATES * 4;                                   // 4096
    constexpr size_t SZ_WVH  = NHID * 4;                                    // 1024
    constexpr size_t SZ_UVC  = 256;
    constexpr size_t SZ_ESUM = ((size_t)BTILES * TSTEPS * 4 + 255) & ~255ull;

    size_t off = 0;
    _Float16* whhP = (_Float16*)(ws + off); off += SZ_WHH;
    _Float16* wihP = (_Float16*)(ws + off); off += SZ_WIH;
    _Float16* xP   = (_Float16*)(ws + off); off += SZ_XP;
    float* bias = (float*)(ws + off); off += SZ_BIAS;
    float* wvh  = (float*)(ws + off); off += SZ_WVH;
    float* uvc  = (float*)(ws + off); off += SZ_UVC;
    float* esum = (float*)(ws + off); off += SZ_ESUM;
    float* hsum = (float*)(ws + off);

    k_pack_whh<<<(NTG * KTILES * 32 + 255) / 256, 256, 0, stream>>>(Whh, whhP);
    k_pack_wih<<<(NTG * 32 + 255) / 256, 256, 0, stream>>>(Wih, wihP);
    k_pack_x<<<(BTILES * TSTEPS * 32 + 255) / 256, 256, 0, stream>>>(x, xP);
    k_misc<<<(GATES + NHID + 255) / 256, 256, 0, stream>>>(bih, bhh, wvW, wr, wc, wvb,
                                                           bias, wvh, uvc);
    k_lstm_attn<<<BTILES, 256, 0, stream>>>(whhP, wihP, xP, bias, wvh, uvc, esum, hsum);
    k_head<<<TSTEPS, 256, 0, stream>>>(esum, hsum, W1, b1, W2, b2, out);
}
